// CrossAttentionFusionBlock_30855045054935
// MI455X (gfx1250) — matrix operand order repaired and checked by reading, not compiled, not run on hardware
//
#include <hip/hip_runtime.h>
#include <hip/hip_bf16.h>

// CrossAttentionFusionBlock for MI455X (gfx1250, wave32, WMMA bf16 16x16x32)
// B=8, C=96, H=W=256, HW=65536, HEADS=4, head_dim=24

#define HW    65536
#define NB    8
#define NC    96
#define HEADS 4
#define HD    24

#define USE_ASYNC_CONV 1   // global_load_async_to_lds_b128 staging in conv3x3

typedef unsigned short u16;
typedef unsigned int   u32;
typedef __attribute__((ext_vector_type(8)))  u16    u16x8;
typedef __attribute__((ext_vector_type(16))) __bf16 v16bf;
typedef __attribute__((ext_vector_type(8)))  float  v8f;

union Frag { v16bf v; u16x8 h[2]; };

__device__ __forceinline__ u16 f2bf(float f) {
  union { float f; u32 u; } x; x.f = f;
  u32 u = x.u;
  u32 r = u + 0x7FFFu + ((u >> 16) & 1u);   // round-to-nearest-even
  return (u16)(r >> 16);
}
__device__ __forceinline__ float bf2f(u16 h) {
  union { u32 u; float f; } x; x.u = ((u32)h) << 16;
  return x.f;
}

// ---------------------------------------------------------------- weights f32->bf16
__global__ __launch_bounds__(256)
void cvt_w(const float* __restrict__ src, u16* __restrict__ dst, int n) {
  int i = blockIdx.x * 256 + threadIdx.x;
  if (i < n) dst[i] = f2bf(src[i]);
}

// ---------------------------------------------------------------- 1x1 conv = GEMM, K=96
// Y[b][oc][p] = sum_k W[oc][k] * X[b][k][p].  Block: 8 waves, each a 16x16 WMMA tile.
// X tile staged via LDS transposed to K-major; each thread stores 8 consecutive k
// for one pixel => single ds_store_b128 per iteration (vs scattered b16 stores).
template<bool IN_F32, bool OUT_F32>
__global__ __launch_bounds__(256)
void gemm96(const void* __restrict__ Xv, const u16* __restrict__ W,
            void* __restrict__ Yv, int OC) {
  __shared__ u16 Xs[128][104];               // [pixel][k], padded stride
  const int tid = threadIdx.x;
  const int b   = blockIdx.z;
  const int oc0 = blockIdx.y * 16;
  const int p0  = blockIdx.x * 128;

  const int sp  = tid & 127;                 // pixel this thread stages
  const int sko = (tid >> 7) * 8;            // k-octet select (0 or 8)
  if (IN_F32) {
    const float* X = (const float*)Xv;
    for (int it = 0; it < 6; ++it) {
      int k0 = it * 16 + sko;
      u16x8 t;
#pragma unroll
      for (int j = 0; j < 8; ++j)
        t[j] = f2bf(X[((size_t)b * 96 + k0 + j) * HW + p0 + sp]);
      *(u16x8*)&Xs[sp][k0] = t;
    }
  } else {
    const u16* X = (const u16*)Xv;
    for (int it = 0; it < 6; ++it) {
      int k0 = it * 16 + sko;
      u16x8 t;
#pragma unroll
      for (int j = 0; j < 8; ++j)
        t[j] = X[((size_t)b * 96 + k0 + j) * HW + p0 + sp];
      *(u16x8*)&Xs[sp][k0] = t;
    }
  }
  __syncthreads();

  const int wv = tid >> 5, lane = tid & 31;
  const int lrow = lane & 15, hi = lane >> 4;
  const int pl = wv * 16 + lrow;             // pixel column of this lane's B frag

  v8f acc = {};
  for (int kc = 0; kc < 96; kc += 32) {
    Frag a, bf;
    // A: row = oc0+lrow; K = kc + hi*16 .. +15 contiguous
    const u16* wr = W + (size_t)(oc0 + lrow) * 96 + kc + hi * 16;
    a.h[0] = *(const u16x8*)(wr);
    a.h[1] = *(const u16x8*)(wr + 8);
    // B: col = lrow; K = kc + hi*16 .. +15 contiguous in LDS
    const u16* xr = &Xs[pl][kc + hi * 16];
    bf.h[0] = *(const u16x8*)(xr);
    bf.h[1] = *(const u16x8*)(xr + 8);
    acc = __builtin_amdgcn_wmma_f32_16x16x32_bf16(false, a.v, false, bf.v,
                                                  (short)0, acc, false, false);
  }

  const int col = p0 + wv * 16 + lrow;
  if (OUT_F32) {
    float* Y = (float*)Yv;
    for (int r = 0; r < 8; ++r)
      Y[((size_t)b * OC + (oc0 + r + hi * 8)) * HW + col] = acc[r];
  } else {
    u16* Y = (u16*)Yv;
    for (int r = 0; r < 8; ++r)
      Y[((size_t)b * OC + (oc0 + r + hi * 8)) * HW + col] = f2bf(acc[r]);
  }
}

// ---------------------------------------------------------------- grouped 3x3 conv, pad=1
// One block per (image row, out-channel, batch). The 3-row halo of the group's
// input channels is staged into LDS with GLOBAL_LOAD_ASYNC_TO_LDS_B128 (ASYNCcnt).
__global__ __launch_bounds__(256)
void conv3x3(const u16* __restrict__ in, const float* __restrict__ w,
             u16* __restrict__ out, int CO, int IPG, int OPG) {
  __shared__ u16 T[2 * 3 * 256];             // [i][r][x], supports IPG<=2
  const int tid = threadIdx.x;
  const int y = blockIdx.x;
  const int o = blockIdx.y, b = blockIdx.z;
  const int g = o / OPG;
  const int CI = (CO / OPG) * IPG;

  const int nchunk = IPG * 96;               // 16B chunks: IPG*3*256*2 / 16
  if (tid < nchunk) {
    int i  = tid / 96;                       // input channel within group
    int cc = tid % 96;
    int rr = cc >> 5;                        // halo row 0..2 -> y-1+rr
    int cx = (cc & 31) * 8;                  // 8 bf16 = 16B per chunk
    int yy = y - 1 + rr;
    u16* lp = &T[(i * 3 + rr) * 256 + cx];
    if (yy >= 0 && yy <= 255) {
      const u16* gp = in + ((size_t)b * CI + g * IPG + i) * HW + yy * 256 + cx;
#if USE_ASYNC_CONV
      u32 lds_off = (u32)(size_t)lp;         // low 32 bits of generic = LDS offset
      asm volatile("global_load_async_to_lds_b128 %0, %1, off"
                   :: "v"(lds_off), "v"(gp) : "memory");
#else
      *(u16x8*)lp = *(const u16x8*)gp;
#endif
    } else {
      u16x8 zz = {};
      *(u16x8*)lp = zz;                      // zero padding rows
    }
  }
#if USE_ASYNC_CONV
  asm volatile("s_wait_asynccnt 0x0" ::: "memory");
#endif
  __syncthreads();

  const int x = tid;
  float acc = 0.f;
  for (int i = 0; i < IPG; ++i) {
    const float* wp = w + ((size_t)o * IPG + i) * 9;
    for (int r = 0; r < 3; ++r) {
      const u16* row = &T[(i * 3 + r) * 256];
#pragma unroll
      for (int dx = 0; dx < 3; ++dx) {
        int xx = x + dx - 1;
        if (xx < 0 || xx > 255) continue;
        acc += wp[r * 3 + dx] * bf2f(row[xx]);
      }
    }
  }
  out[((size_t)b * CO + o) * HW + y * 256 + x] = f2bf(acc);
}

// ---------------------------------------------------------------- per-row L2 norms
__global__ __launch_bounds__(256)
void rownorm(const u16* __restrict__ base, float* __restrict__ out, int rpb) {
  __shared__ float red[256];
  int c = blockIdx.x, bi = blockIdx.y, tid = threadIdx.x;
  const u16* p = base + ((size_t)bi * rpb + c) * HW;
  float s = 0.f;
  for (int i = tid; i < HW; i += 256) { float v = bf2f(p[i]); s += v * v; }
  red[tid] = s; __syncthreads();
  for (int o = 128; o > 0; o >>= 1) {
    if (tid < o) red[tid] += red[tid + o];
    __syncthreads();
  }
  if (tid == 0) out[bi * 96 + c] = fmaxf(sqrtf(red[0]), 1e-12f);
}

// ---------------------------------------------------------------- S = q . k^T  (K=65536)
// One block per (b,h), 16 waves split K; each wave holds the full 32x32 (padded) tile.
__global__ __launch_bounds__(512)
void qk_gemm(const u16* __restrict__ qb, const u16* __restrict__ kvb,
             float* __restrict__ S) {
  __shared__ float red[32][32];
  int bh = blockIdx.x, b = bh >> 2, h = bh & 3;
  int tid = threadIdx.x, wv = tid >> 5, lane = tid & 31;
  int lrow = lane & 15, hi = lane >> 4;
  const u16* qbase = qb  + ((size_t)b * 96  + h * HD) * HW;
  const u16* kbase = kvb + ((size_t)b * 192 + h * HD) * HW;   // k = kv ch 0..95

  v8f acc[2][2] = {};
  int k0 = wv * (HW / 16);
  for (int kc = k0; kc < k0 + HW / 16; kc += 32) {
    Frag a[2], bb[2];
    for (int mt = 0; mt < 2; ++mt) {
      const u16* r = qbase + (size_t)(mt * 16 + lrow) * HW + kc + hi * 16;
      a[mt].h[0] = *(const u16x8*)r;
      a[mt].h[1] = *(const u16x8*)(r + 8);
    }
    for (int nt = 0; nt < 2; ++nt) {
      const u16* r = kbase + (size_t)(nt * 16 + lrow) * HW + kc + hi * 16;
      bb[nt].h[0] = *(const u16x8*)r;
      bb[nt].h[1] = *(const u16x8*)(r + 8);
    }
    for (int mt = 0; mt < 2; ++mt)
      for (int nt = 0; nt < 2; ++nt)
        acc[mt][nt] = __builtin_amdgcn_wmma_f32_16x16x32_bf16(
            false, a[mt].v, false, bb[nt].v, (short)0, acc[mt][nt], false, false);
  }
  for (int i = tid; i < 1024; i += 512) red[i >> 5][i & 31] = 0.f;
  __syncthreads();
  for (int mt = 0; mt < 2; ++mt)
    for (int nt = 0; nt < 2; ++nt)
      for (int r = 0; r < 8; ++r)
        atomicAdd(&red[mt * 16 + hi * 8 + r][nt * 16 + lrow], acc[mt][nt][r]);
  __syncthreads();
  for (int i = tid; i < 1024; i += 512)
    S[(size_t)bh * 1024 + i] = red[i >> 5][i & 31];
}

// ---------------------------------------------------------------- softmax (norms+temp folded)
__global__ __launch_bounds__(32)
void softmax_k(const float* __restrict__ S, const float* __restrict__ nq,
               const float* __restrict__ nk, const float* __restrict__ temp,
               u16* __restrict__ attn) {
  int bh = blockIdx.x, b = bh >> 2, h = bh & 3;
  int c = threadIdx.x;
  u16* arow = attn + (size_t)bh * 1024 + c * 32;
  if (c >= HD) { for (int d = 0; d < 32; ++d) arow[d] = 0; return; }
  float t = temp[h];
  float qn = nq[b * 96 + h * HD + c];
  float lg[HD];
  float mx = -1e30f;
  for (int d = 0; d < HD; ++d) {
    float kn = nk[b * 96 + h * HD + d];
    float l = S[(size_t)bh * 1024 + c * 32 + d] * t / (qn * kn);
    lg[d] = l; mx = fmaxf(mx, l);
  }
  float sum = 0.f;
  for (int d = 0; d < HD; ++d) { lg[d] = __expf(lg[d] - mx); sum += lg[d]; }
  float inv = 1.f / sum;
  for (int d = 0; d < HD; ++d) arow[d] = f2bf(lg[d] * inv);
  for (int d = HD; d < 32; ++d) arow[d] = 0;
}

// ---------------------------------------------------------------- out = attn . v (K padded to 32)
__global__ __launch_bounds__(256)
void av_gemm(const u16* __restrict__ attn, const u16* __restrict__ kvb,
             u16* __restrict__ outb) {
  __shared__ u16 Vs[128][40];                 // [pixel][k], padded stride
  int p0 = blockIdx.x * 128;
  int bh = blockIdx.y, b = bh >> 2, h = bh & 3;
  int tid = threadIdx.x, wv = tid >> 5, lane = tid & 31;
  int lrow = lane & 15, hi = lane >> 4;
  const u16* vbase = kvb + ((size_t)b * 192 + 96 + h * HD) * HW;  // v = kv ch 96..191

  // stage 32 k-rows x 128 pixels transposed; each thread: 8 consecutive k, one pixel
  {
    int sp  = tid & 127;
    int k0  = (tid >> 7) * 8;                 // 0 or 8
    for (int it = 0; it < 2; ++it, k0 += 16) {
      u16x8 t;
#pragma unroll
      for (int j = 0; j < 8; ++j)
        t[j] = vbase[(size_t)(k0 + j) * HW + p0 + sp];  // rows>=24 hit zero attn cols
      *(u16x8*)&Vs[sp][k0] = t;
    }
  }
  __syncthreads();

  const u16* abase = attn + (size_t)bh * 1024;
  Frag bf;
  const u16* xr = &Vs[wv * 16 + lrow][hi * 16];
  bf.h[0] = *(const u16x8*)xr;
  bf.h[1] = *(const u16x8*)(xr + 8);

  v8f acc[2] = {};
  for (int mt = 0; mt < 2; ++mt) {
    Frag a;
    const u16* ar = abase + (size_t)(mt * 16 + lrow) * 32 + hi * 16;
    a.h[0] = *(const u16x8*)ar;
    a.h[1] = *(const u16x8*)(ar + 8);
    acc[mt] = __builtin_amdgcn_wmma_f32_16x16x32_bf16(false, a.v, false, bf.v,
                                                      (short)0, acc[mt], false, false);
  }
  int col = p0 + wv * 16 + lrow;
  for (int mt = 0; mt < 2; ++mt)
    for (int r = 0; r < 8; ++r) {
      int m = mt * 16 + hi * 8 + r;
      if (m < HD)
        outb[((size_t)b * 96 + h * HD + m) * HW + col] = f2bf(acc[mt][r]);
    }
}

// ---------------------------------------------------------------- launch
extern "C" void kernel_launch(void* const* d_in, const int* in_sizes, int n_in,
                              void* d_out, int out_size, void* d_ws, size_t ws_size,
                              hipStream_t stream) {
  const float* x      = (const float*)d_in[0];
  const float* z      = (const float*)d_in[1];
  const float* kv_w1  = (const float*)d_in[2];
  const float* kv_w2  = (const float*)d_in[3];
  const float* q_w1   = (const float*)d_in[4];
  const float* q_w2   = (const float*)d_in[5];
  const float* proj_w = (const float*)d_in[6];
  const float* temp   = (const float*)d_in[7];

  char* ws = (char*)d_ws;
  const size_t SZC = (size_t)NB * 96 * HW * 2;   // one 96-ch bf16 tensor = 100.7 MB
  u16*   wkv1    = (u16*)(ws + 0);               // 192*96 bf16
  u16*   wq1     = (u16*)(ws + 36864);           // 96*96
  u16*   wproj   = (u16*)(ws + 55296);           // 96*96
  float* norms_q = (float*)(ws + 73728);         // 768 f32
  float* norms_k = (float*)(ws + 76800);         // 768 f32
  float* S       = (float*)(ws + 79872);         // 32*32*32 f32
  u16*   attn    = (u16*)(ws + 210944);          // 32*32*32 bf16 (padded)
  const size_t big = (size_t)1 << 20;
  u16* kv1  = (u16*)(ws + big);                  // 2*SZC  (kv after 1x1)
  u16* q1   = (u16*)(ws + big + 2 * SZC);        // SZC    (q after 1x1)
  u16* kv   = (u16*)(ws + big + 3 * SZC);        // 2*SZC  (kv after grouped 3x3)
  u16* qb   = (u16*)(ws + big + 5 * SZC);        // SZC    (q after depthwise 3x3)
  u16* aout = kv1;                               // reuse kv1 region for attn output
  // total needed: 1MB + 6*SZC + 1MB slack for padded-tile over-reads ~= 606 MB

  cvt_w<<<72, 256, 0, stream>>>(kv_w1, wkv1, 192 * 96);
  cvt_w<<<36, 256, 0, stream>>>(q_w1, wq1, 96 * 96);
  cvt_w<<<36, 256, 0, stream>>>(proj_w, wproj, 96 * 96);

  gemm96<true, false><<<dim3(HW / 128, 12, NB), 256, 0, stream>>>(z, wkv1, kv1, 192);
  gemm96<true, false><<<dim3(HW / 128, 6, NB), 256, 0, stream>>>(x, wq1, q1, 96);

  conv3x3<<<dim3(256, 192, NB), 256, 0, stream>>>(kv1, kv_w2, kv, 192, 2, 2);
  conv3x3<<<dim3(256, 96, NB), 256, 0, stream>>>(q1, q_w2, qb, 96, 1, 1);

  rownorm<<<dim3(96, NB), 256, 0, stream>>>(qb, norms_q, 96);
  rownorm<<<dim3(96, NB), 256, 0, stream>>>(kv, norms_k, 192);

  qk_gemm<<<NB * HEADS, 512, 0, stream>>>(qb, kv, S);
  softmax_k<<<NB * HEADS, 32, 0, stream>>>(S, norms_q, norms_k, temp, attn);
  av_gemm<<<dim3(HW / 128, NB * HEADS), 256, 0, stream>>>(attn, kv, aout);

  gemm96<false, true><<<dim3(HW / 128, 6, NB), 256, 0, stream>>>(aout, wproj,
                                                                 (float*)d_out, 96);
}
